// RBLNGptOssMLP_18442589570204
// MI455X (gfx1250) — compile-verified, matrix operand order checked by reading, not run on hardware
//
#include <hip/hip_runtime.h>
#include <hip/hip_bf16.h>
#include <stdint.h>

// ---------------------------------------------------------------------------
// GPT-OSS MoE MLP forward for MI455X (gfx1250, wave32, WMMA)
//   E=16 experts, H=1024, I=2048, top-k=4, MXFP4 expert weights.
// Round 3: replace IEEE division in the clamped-swiglu with v_rcp_f32
// (__builtin_amdgcn_rcpf) -- one TRANS op instead of a ~13-instruction
// div_scale/div_fmas/div_fixup chain per element in the tile epilogue.
// Dequant stays on the v_perm_b32 byte-LUT path feeding v_wmma_f32_16x16x32_f16.
// ---------------------------------------------------------------------------

typedef __attribute__((ext_vector_type(16))) _Float16 v16h;
typedef __attribute__((ext_vector_type(8)))  _Float16 v8h;
typedef __attribute__((ext_vector_type(2)))  _Float16 h2v;
typedef __attribute__((ext_vector_type(8)))  float    v8f;

#define T_TOK 1024
#define H_DIM 1024
#define I_DIM 2048
#define N_EXP 16
#define TOPK  4
#define ALPHA_C 1.702f
#define LIMIT_C 7.0f

union ABu {
  v16h v;
  v8h  h2[2];
  uint4 q[2];
};

__device__ __forceinline__ v8f wmma_f16(v16h a, v16h b, v8f c) {
  // D = A(16x32 f16) * B(32x16 f16) + C(16x16 f32)
  return __builtin_amdgcn_wmma_f32_16x16x32_f16(
      /*neg_a=*/false, a, /*neg_b=*/false, b,
      /*c_mod=*/(short)0, c, /*reuse_a=*/false, /*reuse_b=*/false);
}

// FP16 high-byte magnitude table for FP4 E2M1 codes 0..7:
//   {0, 0.5, 1, 1.5, 2, 3, 4, 6} -> {0x00,0x38,0x3C,0x3E,0x40,0x42,0x44,0x46}
#define FP4_TBL_HI 0x46444240u  // bytes 4..7 (S0 of v_perm)
#define FP4_TBL_LO 0x3E3C3800u  // bytes 0..3 (S1 of v_perm)

// Build broadcast half2 scale = 2^(sb-127) via direct exponent construction.
__device__ __forceinline__ h2v scale_h2(uint32_t sb) {
  int eh = (int)sb - 112;                      // f16 biased exponent
  eh = eh < 1 ? 1 : (eh > 30 ? 30 : eh);       // clamp to normal f16 range
  union { uint16_t b; _Float16 h; } c;
  c.b = (uint16_t)(eh << 10);
  h2v s;
  s[0] = c.h;
  s[1] = c.h;
  return s;
}

// Dequant 8 packed FP4 nibbles (one dword) -> 8 scaled halfs using v_perm LUT.
// Even nibbles -> halfs 0,2,4,6 ; odd nibbles -> halfs 1,3,5,7.
__device__ __forceinline__ v8h dq8(uint32_t w, h2v s2) {
  uint32_t selE = w & 0x07070707u;
  uint32_t selO = (w >> 4) & 0x07070707u;
  uint32_t magE = __builtin_amdgcn_perm(FP4_TBL_HI, FP4_TBL_LO, selE);
  uint32_t magO = __builtin_amdgcn_perm(FP4_TBL_HI, FP4_TBL_LO, selO);
  uint32_t bE = magE | ((w & 0x08080808u) << 4);  // even sign bytes
  uint32_t bO = magO | (w & 0x80808080u);         // odd sign bytes (in place)
  union { uint32_t u[4]; v8h v; h2v h[4]; } r;
  // Interleave with zero low bytes: dword j = {0, E_j, 0, O_j} (sel 0x0C = 0).
  r.u[0] = __builtin_amdgcn_perm(bE, bO, 0x000C040Cu);
  r.u[1] = __builtin_amdgcn_perm(bE, bO, 0x010C050Cu);
  r.u[2] = __builtin_amdgcn_perm(bE, bO, 0x020C060Cu);
  r.u[3] = __builtin_amdgcn_perm(bE, bO, 0x030C070Cu);
#pragma unroll
  for (int i = 0; i < 4; ++i) r.h[i] *= s2;       // v_pk_mul_f16 by 2^(e-127)
  return r.v;
}

// Load one B-tile column's worth of MXFP4 weights for a 32-wide K step.
// Lane holds column n; halves 0..7 = K kb..kb+7, halves 8..15 = K kb+16..kb+23.
__device__ __forceinline__ v16h load_b_q4(const uint8_t* __restrict__ row_blk,
                                          const uint8_t* __restrict__ row_scl,
                                          int k0, int kboff) {
  h2v s2 = scale_h2(row_scl[k0 >> 5]);
  uint32_t w0 = *(const uint32_t*)(row_blk + ((k0 + kboff) >> 1));
  uint32_t w1 = *(const uint32_t*)(row_blk + ((k0 + kboff + 16) >> 1));
  ABu b;
  b.h2[0] = dq8(w0, s2);
  b.h2[1] = dq8(w1, s2);
  return b.v;
}

// ---------------------------------------------------------------------------
__global__ void zero_init_kernel(float* __restrict__ out, int n,
                                 int* __restrict__ cnt) {
  int i = blockIdx.x * blockDim.x + threadIdx.x;
  if (i < n) out[i] = 0.0f;
  if (i < N_EXP) cnt[i] = 0;
}

__global__ void cvt_kernel(const float* __restrict__ x,
                           _Float16* __restrict__ xh, int n) {
  int i = blockIdx.x * blockDim.x + threadIdx.x;
  if (i < n) xh[i] = (_Float16)x[i];
}

// ---------------------------------------------------------------------------
// Router: one wave (32 lanes) per token. 16 logits, top-4 softmax, gather.
__global__ __launch_bounds__(256)
void router_kernel(const float* __restrict__ x, const float* __restrict__ rw,
                   const float* __restrict__ rb, int* __restrict__ cnt,
                   int* __restrict__ tok, float* __restrict__ wtl) {
  const int t = blockIdx.x * 8 + (threadIdx.x >> 5);
  const int lane = threadIdx.x & 31;
  const float* xr = x + (size_t)t * H_DIM;
  float lg[N_EXP];
#pragma unroll 1
  for (int e = 0; e < N_EXP; ++e) {
    const float* wr = rw + (size_t)e * H_DIM;
    float s = 0.0f;
    for (int h = lane; h < H_DIM; h += 32) s = fmaf(xr[h], wr[h], s);
#pragma unroll
    for (int off = 16; off > 0; off >>= 1) s += __shfl_xor(s, off, 32);
    lg[e] = s + rb[e];
  }
  if (lane == 0) {
    int   idx[TOPK];
    float val[TOPK];
#pragma unroll
    for (int k = 0; k < TOPK; ++k) {
      int bi = 0;
      float bv = -3.4e38f;
#pragma unroll
      for (int e = 0; e < N_EXP; ++e)
        if (lg[e] > bv) { bv = lg[e]; bi = e; }
      idx[k] = bi;
      val[k] = bv;
      lg[bi] = -3.4e38f;
    }
    float mx = val[0];
    float p[TOPK];
    float se = 0.0f;
#pragma unroll
    for (int k = 0; k < TOPK; ++k) { p[k] = __expf(val[k] - mx); se += p[k]; }
    float inv = __builtin_amdgcn_rcpf(se);  // v_rcp_f32
#pragma unroll
    for (int k = 0; k < TOPK; ++k) {
      int pos = atomicAdd(&cnt[idx[k]], 1);
      tok[idx[k] * T_TOK + pos] = t;
      wtl[idx[k] * T_TOK + pos] = p[k] * inv;
    }
  }
}

// ---------------------------------------------------------------------------
// Fused expert block: blockIdx.y = expert, blockIdx.x = 16-token tile.
// 8 waves. LDS: x tile (16x1024 f16) + inter chunk (16x512 f16).
__global__ __launch_bounds__(256)
void moe_expert_kernel(const _Float16* __restrict__ xh,
                       const uint8_t* __restrict__ gblk,
                       const uint8_t* __restrict__ gscl,
                       const uint8_t* __restrict__ ublk,
                       const uint8_t* __restrict__ uscl,
                       const uint8_t* __restrict__ dblk,
                       const uint8_t* __restrict__ dscl,
                       const float* __restrict__ gbias,
                       const float* __restrict__ ubias,
                       const float* __restrict__ dbias,
                       const int* __restrict__ cnt,
                       const int* __restrict__ tok,
                       const float* __restrict__ wtl,
                       float* __restrict__ out) {
  const int e = blockIdx.y;
  const int c = cnt[e];
  const int base = blockIdx.x * 16;
  if (base >= c) return;  // uniform exit: tile beyond this expert's token count

  __shared__ __align__(16) _Float16 sx[16][1032];  // +8 halfs pad, 16B aligned
  __shared__ __align__(16) _Float16 si[16][520];   // 512 + 8 pad
  __shared__ int   stok[16];
  __shared__ float swt[16];

  const int tid  = threadIdx.x;
  const int wave = tid >> 5;
  const int lane = tid & 31;
  const int nlo  = lane & 15;          // row (A) / column (B,D) index
  const int khalf = (lane >> 4) & 1;   // K-half select per WMMA layout
  const int kboff = khalf * 8;

  if (tid < 16) {
    int slot = base + tid;
    int cs = slot < c ? slot : (c - 1);
    stok[tid] = tok[e * T_TOK + cs];
    swt[tid]  = slot < c ? wtl[e * T_TOK + slot] : 0.0f;
  }
  __syncthreads();

  // Stage activation tile: 16 tokens x 1024 halfs, 128B per thread-chunk.
  {
    int r = tid >> 4;
    int cc = tid & 15;
    const _Float16* src = xh + (size_t)stok[r] * H_DIM + cc * 64;
    _Float16* dst = &sx[r][cc * 64];
#pragma unroll
    for (int i = 0; i < 8; ++i)
      ((uint4*)dst)[i] = ((const uint4*)src)[i];
  }
  __syncthreads();

  // Persistent down-proj accumulators: wave owns 8 h-tiles (128 columns).
  v8f accd[8];
#pragma unroll
  for (int j = 0; j < 8; ++j) accd[j] = (v8f){};

#pragma unroll 1
  for (int cb = 0; cb < 4; ++cb) {      // I chunks of 512
    const int ichunk = cb * 512;

    // ---- Phase 1: gate/up for this chunk (32 n-tiles, 4 per wave) ----
#pragma unroll 1
    for (int q = 0; q < 4; ++q) {
      const int ntl = wave * 64 + q * 16;        // chunk-local column base
      const int icol = ichunk + ntl + nlo;       // global intermediate channel
      const uint8_t* grow  = gblk + (size_t)(e * I_DIM + icol) * (H_DIM / 2);
      const uint8_t* gsrow = gscl + (size_t)(e * I_DIM + icol) * (H_DIM / 32);
      const uint8_t* urow  = ublk + (size_t)(e * I_DIM + icol) * (H_DIM / 2);
      const uint8_t* usrow = uscl + (size_t)(e * I_DIM + icol) * (H_DIM / 32);
      // Stream hint for the 512B MXFP4 rows (global_prefetch_b8).
      __builtin_prefetch(grow, 0, 3);
      __builtin_prefetch(grow + 256, 0, 3);
      __builtin_prefetch(urow, 0, 3);
      __builtin_prefetch(urow + 256, 0, 3);
      v8f accg = (v8f){};
      v8f accu = (v8f){};
#pragma unroll 1
      for (int k0 = 0; k0 < H_DIM; k0 += 32) {
        ABu a;
        a.q[0] = *(const uint4*)&sx[nlo][k0 + kboff];
        a.q[1] = *(const uint4*)&sx[nlo][k0 + kboff + 16];
        v16h bg = load_b_q4(grow, gsrow, k0, kboff);
        accg = wmma_f16(a.v, bg, accg);
        v16h bu = load_b_q4(urow, usrow, k0, kboff);
        accu = wmma_f16(a.v, bu, accu);
      }
      const float gb = gbias[e * I_DIM + icol];
      const float ub = ubias[e * I_DIM + icol];
#pragma unroll
      for (int r = 0; r < 8; ++r) {
        // D layout: VGPR r -> M = r (+8 for upper half-wave), N = nlo.
        int m = r + khalf * 8;
        float g = accg[r] + gb;
        float u = accu[r] + ub;
        g = fminf(g, LIMIT_C);
        u = fminf(fmaxf(u, -LIMIT_C), LIMIT_C);
        // g * sigmoid(a*g) with fast reciprocal (v_rcp_f32, 1 TRANS op)
        float glu = g * __builtin_amdgcn_rcpf(1.0f + __expf(-ALPHA_C * g));
        si[m][ntl + nlo] = (_Float16)((u + 1.0f) * glu);
      }
    }
    __syncthreads();

    // ---- Phase 2: down-proj partial over this chunk ----
#pragma unroll 1
    for (int k0 = 0; k0 < 512; k0 += 32) {
      ABu a;
      a.q[0] = *(const uint4*)&si[nlo][k0 + kboff];
      a.q[1] = *(const uint4*)&si[nlo][k0 + kboff + 16];
      const int kg = ichunk + k0;  // global K (intermediate dim) base
#pragma unroll
      for (int j = 0; j < 8; ++j) {
        int h = wave * 128 + j * 16 + nlo;
        const uint8_t* drow  = dblk + (size_t)(e * H_DIM + h) * (I_DIM / 2);
        const uint8_t* dsrow = dscl + (size_t)(e * H_DIM + h) * (I_DIM / 32);
        v16h bd = load_b_q4(drow, dsrow, kg, kboff);
        accd[j] = wmma_f16(a.v, bd, accd[j]);
      }
    }
    __syncthreads();
  }

  // ---- Epilogue: routing-weighted accumulate into output ----
#pragma unroll
  for (int j = 0; j < 8; ++j) {
    const int h = wave * 128 + j * 16 + nlo;
    const float db = dbias[e * H_DIM + h];
#pragma unroll
    for (int r = 0; r < 8; ++r) {
      int m = r + khalf * 8;
      float w = swt[m];
      if (w != 0.0f) {
        atomicAdd(&out[(size_t)stok[m] * H_DIM + h], w * (accd[j][r] + db));
      }
    }
  }
}

// ---------------------------------------------------------------------------
extern "C" void kernel_launch(void* const* d_in, const int* in_sizes, int n_in,
                              void* d_out, int out_size, void* d_ws,
                              size_t ws_size, hipStream_t stream) {
  (void)in_sizes; (void)n_in; (void)out_size; (void)ws_size;
  const float* x    = (const float*)d_in[0];
  const float* rw   = (const float*)d_in[1];
  const float* rb   = (const float*)d_in[2];
  const float* gb   = (const float*)d_in[3];
  const float* ub   = (const float*)d_in[4];
  const float* db   = (const float*)d_in[5];
  const uint8_t* gblk = (const uint8_t*)d_in[6];
  const uint8_t* ublk = (const uint8_t*)d_in[7];
  const uint8_t* dblk = (const uint8_t*)d_in[8];
  const uint8_t* gscl = (const uint8_t*)d_in[9];
  const uint8_t* uscl = (const uint8_t*)d_in[10];
  const uint8_t* dscl = (const uint8_t*)d_in[11];
  float* out = (float*)d_out;

  char* ws = (char*)d_ws;
  _Float16* xh = (_Float16*)ws;                                  // 2 MB
  int*   cnt = (int*)(ws + (size_t)T_TOK * H_DIM * 2);           // 64 B
  int*   tok = (int*)((char*)cnt + 256);                         // 64 KB
  float* wtl = (float*)((char*)tok + (size_t)N_EXP * T_TOK * 4); // 64 KB

  const int n = T_TOK * H_DIM;
  zero_init_kernel<<<n / 256, 256, 0, stream>>>(out, n, cnt);
  cvt_kernel<<<n / 256, 256, 0, stream>>>(x, xh, n);
  router_kernel<<<T_TOK / 8, 256, 0, stream>>>(x, rw, rb, cnt, tok, wtl);
  moe_expert_kernel<<<dim3(T_TOK / 16, N_EXP), 256, 0, stream>>>(
      xh, gblk, gscl, ublk, uscl, dblk, dscl, gb, ub, db, cnt, tok, wtl, out);
}
